// Interaction_55439437857312
// MI455X (gfx1250) — matrix operand order repaired and checked by reading, not compiled, not run on hardware
//
#include <hip/hip_runtime.h>
#include <hip/hip_bf16.h>
#include <cstdint>

// ---------------------------------------------------------------------------
// Types / helpers
// ---------------------------------------------------------------------------
typedef _Float16 v16h __attribute__((ext_vector_type(16)));
typedef _Float16 v8h  __attribute__((ext_vector_type(8)));
typedef float    v8f  __attribute__((ext_vector_type(8)));
typedef int      v4i  __attribute__((ext_vector_type(4)));

#define IMG 128          // H == W
#define BATCH 8
#define CIN_TOT 256      // total input channels for both 3x3 convs
#define INTER 128
#define NHEADS 8
#define NWIN 2048        // B * 16 * 16 windows

// gfx1250 async global->LDS copy (ASYNCcnt-tracked), guarded for toolchain.
// Probe-derived signature: (as1 v4i* src_global, as3 v4i* dst_lds, imm, imm).
#if defined(__has_builtin)
#if __has_builtin(__builtin_amdgcn_global_load_async_to_lds_b128)
#define HAVE_ASYNC_LDS 1
#endif
#endif
#ifndef HAVE_ASYNC_LDS
#define HAVE_ASYNC_LDS 0
#endif

#if HAVE_ASYNC_LDS
#define ASYNC_COPY_B128(src, dst)                                              \
  __builtin_amdgcn_global_load_async_to_lds_b128(                              \
      (__attribute__((address_space(1))) v4i*)(src),                           \
      (__attribute__((address_space(3))) v4i*)(dst), 0, 0)
#if __has_builtin(__builtin_amdgcn_s_wait_asynccnt)
#define ASYNC_WAIT() __builtin_amdgcn_s_wait_asynccnt(0)
#else
#define ASYNC_WAIT() asm volatile("s_wait_asynccnt 0x0" ::: "memory")
#endif
#else
#define ASYNC_WAIT()
#endif

__device__ __forceinline__ v8f wmma_f16(v16h a, v16h b, v8f c) {
  return __builtin_amdgcn_wmma_f32_16x16x32_f16(false, a, false, b, (short)0, c,
                                                false, false);
}

// A-fragment (16x32 f16): lane L holds row m=L&15; halfs K in [kb,kb+8) and
// [kb+16,kb+24) with kb=(L>>4)*8.  p points at &A[m][kb].
__device__ __forceinline__ v16h ld_afrag(const _Float16* p) {
  v8h lo = *(const v8h*)p;
  v8h hi = *(const v8h*)(p + 16);
  v16h r;
#pragma unroll
  for (int i = 0; i < 8; ++i) { r[i] = lo[i]; r[i + 8] = hi[i]; }
  return r;
}
// B-fragment (32x16 f16, stored as Bmat[n][k]): lane L holds col n=L&15;
// halfs K in [kB,kB+8) and [kB+16,kB+24) with kB=(L>>4)*8.  p points at &B[n][kB].
__device__ __forceinline__ v16h ld_bfrag(const _Float16* p) {
  v8h lo = *(const v8h*)p;
  v8h hi = *(const v8h*)(p + 16);
  v16h r;
#pragma unroll
  for (int i = 0; i < 8; ++i) { r[i] = lo[i]; r[i + 8] = hi[i]; }
  return r;
}

// ---------------------------------------------------------------------------
// Weight prep: [OC][256][3][3] f32 -> [tap][OC][256] f16
// ---------------------------------------------------------------------------
__global__ void prep_conv_w(const float* __restrict__ w, _Float16* __restrict__ wt,
                            int OC) {
  int idx = blockIdx.x * 256 + threadIdx.x;
  int total = OC * 256 * 9;
  if (idx >= total) return;
  int tap = idx % 9;
  int ic  = (idx / 9) % 256;
  int oc  = idx / (9 * 256);
  wt[((size_t)tap * OC + oc) * 256 + ic] = (_Float16)w[idx];
}

// Pack q/k/v 1x1 weights into [224][128] f16 + fused bias [224] f32
__global__ void prep_qkv_w(const float* __restrict__ wq, const float* __restrict__ bq,
                           const float* __restrict__ wk, const float* __restrict__ bk,
                           const float* __restrict__ wv, const float* __restrict__ bv,
                           _Float16* __restrict__ wout, float* __restrict__ bout) {
  int idx = blockIdx.x * 256 + threadIdx.x;
  if (idx < 224 * 128) {
    int n = idx >> 7, c = idx & 127;
    float v;
    if (n < 32)       v = wq[n * 128 + c];
    else if (n < 96)  v = wk[(n - 32) * 128 + c];
    else              v = wv[(n - 96) * 128 + c];
    wout[idx] = (_Float16)v;
  }
  if (idx < 224) {
    bout[idx] = (idx < 32) ? bq[idx] : (idx < 96) ? bk[idx - 32] : bv[idx - 96];
  }
}

// ---------------------------------------------------------------------------
// Implicit-GEMM 3x3 conv + BN(affine) + ReLU, software-pipelined.
//   grid.x = B*IMG (one image row per WG), grid.z = COUT/128 (oc block)
//   256 threads = 8 waves; wave tiling 2M x 4N (8 accum tiles per wave).
//   K = 9 taps * 256 cin in 72 chunks of 32; ping-pong LDS buffers hide
//   global latency.  B tiles (f16 weights) go through async global->LDS
//   when the toolchain exposes it; A tiles need f32->f16 convert so they
//   pipeline through VGPRs.
// ---------------------------------------------------------------------------
template <bool OUT_F16>
__global__ __launch_bounds__(256) void conv3x3_kernel(
    const float* __restrict__ inA, const float* __restrict__ inB, int cinA,
    const _Float16* __restrict__ wt, const float* __restrict__ gamma,
    const float* __restrict__ beta, void* __restrict__ outp, int COUT) {
  const int by = blockIdx.x;
  const int b = by >> 7;
  const int y = by & 127;
  const int ocBase = blockIdx.z * 128;

  __shared__ _Float16 A_s[2][128 * 40];  // [x][k], stride 40 halfs (80B)
  __shared__ _Float16 B_s[2][128 * 40];  // [oc][k]

  const int tid = threadIdx.x;
  const int wv = tid >> 5;
  const int L  = tid & 31;
  const int mrow = L & 15;
  const int kb = (L >> 4) * 8;
  const int kB = (L >> 4) * 8;
  const int wr = wv >> 1;  // M-tile pair 2wr, 2wr+1
  const int wc = wv & 1;   // N-tiles 4wc .. 4wc+3

  // loader roles
  const int lk = tid & 31;   // A: channel within chunk
  const int xg = tid >> 5;   // A: x-group (16 pixels)
  const int boc = tid >> 2;  // B: oc row (plus +64 on second issue)
  const int bh = tid & 3;    // B: 16B quarter-row

  v8f acc[2][4];
#pragma unroll
  for (int mi = 0; mi < 2; ++mi)
#pragma unroll
    for (int j = 0; j < 4; ++j)
      acc[mi][j] = (v8f){0.f, 0.f, 0.f, 0.f, 0.f, 0.f, 0.f, 0.f};

  // ---- A: global f32 -> regs (chunk `it`) -------------------------------
  auto a_load = [&](int it, float* av) {
    const int tap = it >> 3;
    const int cc = (it & 7) * 32;
    const int dy = tap / 3 - 1, dx = tap % 3 - 1;
    const int iy = y + dy;
    const bool rowok = (iy >= 0) && (iy < IMG);
    const int c = cc + lk;
    const float* row = inA;
    if (rowok) {
      if (c < cinA)
        row = inA + (((size_t)b * cinA + c) * IMG + iy) * IMG;
      else
        row = inB + (((size_t)b * (CIN_TOT - cinA) + (c - cinA)) * IMG + iy) * IMG;
    }
    if (!rowok) {
#pragma unroll
      for (int i = 0; i < 16; ++i) av[i] = 0.f;
    } else if (xg != 0 && xg != 7) {  // interior: no bounds checks
      const float* p = row + xg * 16 + dx;
#pragma unroll
      for (int i = 0; i < 16; ++i) av[i] = p[i];
    } else {
#pragma unroll
      for (int i = 0; i < 16; ++i) {
        const int ix = xg * 16 + i + dx;
        av[i] = (ix >= 0 && ix < IMG) ? row[ix] : 0.f;
      }
    }
  };
  auto a_store = [&](const float* av, _Float16* dst) {
#pragma unroll
    for (int i = 0; i < 16; ++i) dst[(xg * 16 + i) * 40 + lk] = (_Float16)av[i];
  };

  // ---- B: f16 weights -> LDS (async when available) ---------------------
  auto b_src = [&](int it, int e) -> const _Float16* {
    const int tap = it >> 3;
    const int cc = (it & 7) * 32;
    const int oc = boc + e * 64;
    return wt + ((size_t)tap * COUT + ocBase + oc) * 256 + cc + bh * 8;
  };
  auto b_dst = [&](_Float16* buf, int e) -> _Float16* {
    const int oc = boc + e * 64;
    return buf + oc * 40 + bh * 8;
  };
#if HAVE_ASYNC_LDS
  auto b_fill_async = [&](int it, _Float16* buf) {
#pragma unroll
    for (int e = 0; e < 2; ++e) {
      ASYNC_COPY_B128(b_src(it, e), b_dst(buf, e));
    }
  };
#endif

  // ---- prologue: fill buffer 0 with chunk 0 -----------------------------
  {
    float av[16];
    a_load(0, av);
    a_store(av, A_s[0]);
#if HAVE_ASYNC_LDS
    b_fill_async(0, B_s[0]);
#else
    const uint4 b0 = *(const uint4*)b_src(0, 0);
    const uint4 b1 = *(const uint4*)b_src(0, 1);
    *(uint4*)b_dst(B_s[0], 0) = b0;
    *(uint4*)b_dst(B_s[0], 1) = b1;
#endif
  }
  ASYNC_WAIT();
  __syncthreads();

  // ---- main pipelined loop ----------------------------------------------
  for (int it = 0; it < 72; ++it) {
    const int p = it & 1;
    const bool nxt = (it + 1) < 72;
    float av[16];
#if !HAVE_ASYNC_LDS
    uint4 bv0, bv1;
#endif
    if (nxt) {
      a_load(it + 1, av);
#if HAVE_ASYNC_LDS
      b_fill_async(it + 1, B_s[1 - p]);
#else
      bv0 = *(const uint4*)b_src(it + 1, 0);
      bv1 = *(const uint4*)b_src(it + 1, 1);
#endif
    }
    // compute on buffer p
    v16h af0 = ld_afrag(&A_s[p][((2 * wr + 0) * 16 + mrow) * 40 + kb]);
    v16h af1 = ld_afrag(&A_s[p][((2 * wr + 1) * 16 + mrow) * 40 + kb]);
#pragma unroll
    for (int j = 0; j < 4; ++j) {
      const v16h bf = ld_bfrag(&B_s[p][((wc * 4 + j) * 16 + mrow) * 40 + kB]);
      acc[0][j] = wmma_f16(af0, bf, acc[0][j]);
      acc[1][j] = wmma_f16(af1, bf, acc[1][j]);
    }
    if (nxt) {
      a_store(av, A_s[1 - p]);
#if !HAVE_ASYNC_LDS
      *(uint4*)b_dst(B_s[1 - p], 0) = bv0;
      *(uint4*)b_dst(B_s[1 - p], 1) = bv1;
#endif
    }
    ASYNC_WAIT();
    __syncthreads();
  }

  // ---- epilogue: BN affine + ReLU; lane owns 8 consecutive x ------------
#pragma unroll
  for (int mi = 0; mi < 2; ++mi) {
    const int x0 = (2 * wr + mi) * 16 + 8 * (L >> 4);
#pragma unroll
    for (int j = 0; j < 4; ++j) {
      const int oc = ocBase + (wc * 4 + j) * 16 + mrow;
      const float g = gamma[oc], bb = beta[oc];
      float vals[8];
#pragma unroll
      for (int i = 0; i < 8; ++i) vals[i] = fmaxf(acc[mi][j][i] * g + bb, 0.f);
      const size_t base = (((size_t)b * COUT + oc) * IMG + y) * IMG + x0;
      if constexpr (OUT_F16) {
        v8h h;
#pragma unroll
        for (int i = 0; i < 8; ++i) h[i] = (_Float16)vals[i];
        *(v8h*)((_Float16*)outp + base) = h;
      } else {
        float* o = (float*)outp + base;
        float4 f0, f1;
        f0.x = vals[0]; f0.y = vals[1]; f0.z = vals[2]; f0.w = vals[3];
        f1.x = vals[4]; f1.y = vals[5]; f1.z = vals[6]; f1.w = vals[7];
        *(float4*)o = f0;
        *(float4*)(o + 4) = f1;
      }
    }
  }
}

// ---------------------------------------------------------------------------
// Fused q/k/v 1x1 conv per window: [64 x 128] @ [128 x 224] + bias.
// ---------------------------------------------------------------------------
__global__ __launch_bounds__(256) void qkv_kernel(
    const _Float16* __restrict__ xf, const _Float16* __restrict__ wqkv,
    const float* __restrict__ bqkv, _Float16* __restrict__ q,
    _Float16* __restrict__ k, _Float16* __restrict__ v) {
  const int win = blockIdx.x;
  const int b = win >> 8, wy = (win >> 4) & 15, wx = win & 15;

  __shared__ _Float16 A_s[64 * 72];    // [tok][k-chunk(64)+pad]
  __shared__ _Float16 B_s[224 * 72];   // [n][k-chunk(64)+pad]

  const int tid = threadIdx.x;
  const int wv = tid >> 5;
  const int L = tid & 31;
  const int mrow = L & 15;
  const int kb = (L >> 4) * 8;
  const int kB = (L >> 4) * 8;
  const int mt = wv >> 1;
  const int nbase = (wv & 1) * 7;

  v8f acc[7];
#pragma unroll
  for (int j = 0; j < 7; ++j) acc[j] = (v8f){0.f, 0.f, 0.f, 0.f, 0.f, 0.f, 0.f, 0.f};

  for (int cc = 0; cc < 128; cc += 64) {
    __syncthreads();
    {  // A fill: thread t: channel cc+(t&63), two window rows (t>>6)*2 ..+1
      const int c = cc + (tid & 63);
      const int r0 = (tid >> 6) * 2;
      const _Float16* src =
          xf + (((size_t)b * INTER + c) * IMG + wy * 8) * IMG + wx * 8;
#pragma unroll
      for (int rr = 0; rr < 2; ++rr) {
        const int ry = r0 + rr;
        const _Float16* s = src + (size_t)ry * IMG;
#pragma unroll
        for (int i = 0; i < 8; ++i) A_s[(ry * 8 + i) * 72 + (c - cc)] = s[i];
      }
    }
    if (tid < 224) {  // B fill: one 64-half row per thread (128B)
      const uint4* s = (const uint4*)(wqkv + (size_t)tid * 128 + cc);
      uint4* d = (uint4*)(B_s + tid * 72);
#pragma unroll
      for (int i = 0; i < 8; ++i) d[i] = s[i];
    }
    __syncthreads();
#pragma unroll
    for (int ks = 0; ks < 2; ++ks) {
      const v16h af = ld_afrag(&A_s[(mt * 16 + mrow) * 72 + ks * 32 + kb]);
#pragma unroll
      for (int j = 0; j < 7; ++j) {
        const int nt = nbase + j;
        const v16h bf = ld_bfrag(&B_s[(nt * 16 + mrow) * 72 + ks * 32 + kB]);
        acc[j] = wmma_f16(af, bf, acc[j]);
      }
    }
  }

  // Epilogue: bias, f16 convert, scatter to q/k/v layouts
  const int tbase = mt * 16 + 8 * (L >> 4);
#pragma unroll
  for (int j = 0; j < 7; ++j) {
    const int ch = (nbase + j) * 16 + mrow;
    const float bias = bqkv[ch];
    _Float16* dst;
    int stride;
    if (ch < 32) {
      dst = q + (((size_t)win * NHEADS + (ch >> 2)) * 64) * 4 + (ch & 3);
      stride = 4;
    } else if (ch < 96) {
      const int c2 = ch - 32;
      dst = k + (((size_t)win * NHEADS + (c2 >> 3)) * 64) * 8 + (c2 & 7);
      stride = 8;
    } else {
      const int c2 = ch - 96;
      dst = v + (((size_t)win * NHEADS + (c2 >> 4)) * 64) * 16 + (c2 & 15);
      stride = 16;
    }
#pragma unroll
    for (int i = 0; i < 8; ++i)
      dst[(size_t)(tbase + i) * stride] = (_Float16)(acc[j][i] + bias);
  }
}

// ---------------------------------------------------------------------------
// Windowed cross attention: one block (4 waves) per (window, head).
// ---------------------------------------------------------------------------
__global__ __launch_bounds__(128) void attn_kernel(
    const _Float16* __restrict__ q1, const _Float16* __restrict__ k1,
    const _Float16* __restrict__ v1, const _Float16* __restrict__ q2,
    const _Float16* __restrict__ k2, const _Float16* __restrict__ v2,
    float* __restrict__ x1i, float* __restrict__ x2i) {
  const int win = blockIdx.x;
  const int head = blockIdx.y;
  const int b = win >> 8, wy = (win >> 4) & 15, wx = win & 15;

  __shared__ _Float16 Qs[64 * 40];     // [tok][d pad->32(+8)] zero-padded
  __shared__ _Float16 K1s[64 * 40];
  __shared__ _Float16 K2s[64 * 40];
  __shared__ _Float16 V1s[16 * 72];    // V^T: [d][tok]
  __shared__ _Float16 V2s[16 * 72];
  __shared__ _Float16 As[4 * 16 * 72]; // probability strips per wave

  const int tid = threadIdx.x;
  const size_t qb = ((size_t)win * NHEADS + head) * 64 * 4;
  const size_t kbse = ((size_t)win * NHEADS + head) * 64 * 8;
  const size_t vb = ((size_t)win * NHEADS + head) * 64 * 16;

  if (tid < 64) {
    const int t = tid;
#pragma unroll
    for (int i = 0; i < 4; ++i) {
      Qs[t * 40 + i] = q1[qb + t * 4 + i];
      Qs[t * 40 + 4 + i] = q2[qb + t * 4 + i];
    }
#pragma unroll
    for (int i = 0; i < 8; ++i) {
      K1s[t * 40 + i] = k1[kbse + t * 8 + i];
      K2s[t * 40 + i] = k2[kbse + t * 8 + i];
    }
#pragma unroll
    for (int i = 8; i < 32; ++i) {
      Qs[t * 40 + i] = (_Float16)0.f;
      K1s[t * 40 + i] = (_Float16)0.f;
      K2s[t * 40 + i] = (_Float16)0.f;
    }
  } else {
    const int t = tid - 64;  // token
#pragma unroll
    for (int d = 0; d < 16; ++d) {
      V1s[d * 72 + t] = v1[vb + t * 16 + d];
      V2s[d * 72 + t] = v2[vb + t * 16 + d];
    }
  }
  __syncthreads();

  const int wv = tid >> 5;
  const int L = tid & 31;
  const int mrow = L & 15;
  const int kb = (L >> 4) * 8;
  const int kB = (L >> 4) * 8;

  // S strips: M-tile = wv (tokens wv*16..+15), 4 N-tiles of key tokens
  const v16h qa = ld_afrag(&Qs[(wv * 16 + mrow) * 40 + kb]);
  float dmat[4][8];
#pragma unroll
  for (int j = 0; j < 4; ++j) {
    const v8f z = (v8f){0.f, 0.f, 0.f, 0.f, 0.f, 0.f, 0.f, 0.f};
    const v16h bk1 = ld_bfrag(&K1s[(j * 16 + mrow) * 40 + kB]);
    const v16h bk2 = ld_bfrag(&K2s[(j * 16 + mrow) * 40 + kB]);
    const v8f s1 = wmma_f16(qa, bk1, z);
    const v8f s2 = wmma_f16(qa, bk2, z);
#pragma unroll
    for (int i = 0; i < 8; ++i) dmat[j][i] = fabsf(s1[i] - s2[i]);
  }

  // Row softmax: row lives across lanes of the same 16-lane group + 4 N-tiles
#pragma unroll
  for (int i = 0; i < 8; ++i) {
    float m = dmat[0][i];
#pragma unroll
    for (int j = 1; j < 4; ++j) m = fmaxf(m, dmat[j][i]);
    m = fmaxf(m, __shfl_xor(m, 1, 32));
    m = fmaxf(m, __shfl_xor(m, 2, 32));
    m = fmaxf(m, __shfl_xor(m, 4, 32));
    m = fmaxf(m, __shfl_xor(m, 8, 32));
    float sum = 0.f;
#pragma unroll
    for (int j = 0; j < 4; ++j) {
      dmat[j][i] = __expf(dmat[j][i] - m);
      sum += dmat[j][i];
    }
    sum += __shfl_xor(sum, 1, 32);
    sum += __shfl_xor(sum, 2, 32);
    sum += __shfl_xor(sum, 4, 32);
    sum += __shfl_xor(sum, 8, 32);
    const float inv = 1.f / sum;
#pragma unroll
    for (int j = 0; j < 4; ++j) dmat[j][i] *= inv;
  }

  // Stage probabilities (C-layout -> A-layout via LDS)
  _Float16* Aw = As + wv * (16 * 72);
#pragma unroll
  for (int j = 0; j < 4; ++j)
#pragma unroll
    for (int i = 0; i < 8; ++i) {
      const int row = i + 8 * (L >> 4);
      Aw[row * 72 + j * 16 + mrow] = (_Float16)dmat[j][i];
    }
  __syncthreads();

  // O = P @ V  (M=16, N=16, K=64 in 2 steps)
  v8f o1 = (v8f){0.f, 0.f, 0.f, 0.f, 0.f, 0.f, 0.f, 0.f};
  v8f o2 = o1;
#pragma unroll
  for (int ks = 0; ks < 2; ++ks) {
    const v16h pa = ld_afrag(&Aw[mrow * 72 + ks * 32 + kb]);
    const v16h bv1f = ld_bfrag(&V1s[mrow * 72 + ks * 32 + kB]);
    const v16h bv2f = ld_bfrag(&V2s[mrow * 72 + ks * 32 + kB]);
    o1 = wmma_f16(pa, bv1f, o1);
    o2 = wmma_f16(pa, bv2f, o2);
  }

  // Unwindow store: lane owns channel c=head*16+mrow, row y, 8 consecutive x
  const int c = head * 16 + mrow;
  const int y = wy * 8 + 2 * wv + (L >> 4);
  const size_t base = (((size_t)b * INTER + c) * IMG + y) * IMG + wx * 8;
  float4 f0, f1;
  f0.x = o1[0]; f0.y = o1[1]; f0.z = o1[2]; f0.w = o1[3];
  f1.x = o1[4]; f1.y = o1[5]; f1.z = o1[6]; f1.w = o1[7];
  *(float4*)(x1i + base) = f0;
  *(float4*)(x1i + base + 4) = f1;
  f0.x = o2[0]; f0.y = o2[1]; f0.z = o2[2]; f0.w = o2[3];
  f1.x = o2[4]; f1.y = o2[5]; f1.z = o2[6]; f1.w = o2[7];
  *(float4*)(x2i + base) = f0;
  *(float4*)(x2i + base + 4) = f1;
}

// ---------------------------------------------------------------------------
// Launcher
// ---------------------------------------------------------------------------
extern "C" void kernel_launch(void* const* d_in, const int* in_sizes, int n_in,
                              void* d_out, int out_size, void* d_ws, size_t ws_size,
                              hipStream_t stream) {
  (void)in_sizes; (void)n_in; (void)out_size; (void)ws_size;

  const float* x1    = (const float*)d_in[0];
  const float* x2    = (const float*)d_in[1];
  const float* w_ic1 = (const float*)d_in[2];
  const float* g_ic1 = (const float*)d_in[3];
  const float* b_ic1 = (const float*)d_in[4];
  const float* w_ic2 = (const float*)d_in[5];
  const float* g_ic2 = (const float*)d_in[6];
  const float* b_ic2 = (const float*)d_in[7];
  const float* wq1 = (const float*)d_in[8];
  const float* bq1 = (const float*)d_in[9];
  const float* wk1 = (const float*)d_in[10];
  const float* bk1 = (const float*)d_in[11];
  const float* wv1 = (const float*)d_in[12];
  const float* bv1 = (const float*)d_in[13];
  const float* wq2 = (const float*)d_in[14];
  const float* bq2 = (const float*)d_in[15];
  const float* wk2 = (const float*)d_in[16];
  const float* bk2 = (const float*)d_in[17];
  const float* wv2 = (const float*)d_in[18];
  const float* bv2 = (const float*)d_in[19];
  const float* w_cat = (const float*)d_in[20];
  const float* g_cat = (const float*)d_in[21];
  const float* b_cat = (const float*)d_in[22];

  float* out0 = (float*)d_out;                       // x_cat [8,256,128,128]
  float* out1 = out0 + (size_t)8 * 256 * 128 * 128;  // x1_i [8,128,128,128]
  float* out2 = out1 + (size_t)8 * 128 * 128 * 128;  // x2_i

  // Workspace carve-up (256B aligned)
  char* ws = (char*)d_ws;
  size_t off = 0;
  auto carve = [&](size_t bytes) {
    char* p = ws + off;
    off = (off + bytes + 255) & ~(size_t)255;
    return p;
  };
  _Float16* wt1   = (_Float16*)carve((size_t)9 * 128 * 256 * 2);
  _Float16* wt2   = (_Float16*)carve((size_t)9 * 128 * 256 * 2);
  _Float16* wtc   = (_Float16*)carve((size_t)9 * 256 * 256 * 2);
  _Float16* wqkvA = (_Float16*)carve((size_t)224 * 128 * 2);
  _Float16* wqkvB = (_Float16*)carve((size_t)224 * 128 * 2);
  float*    bqkvA = (float*)carve(224 * 4);
  float*    bqkvB = (float*)carve(224 * 4);
  _Float16* x1f   = (_Float16*)carve((size_t)8 * 128 * 128 * 128 * 2);
  _Float16* x2f   = (_Float16*)carve((size_t)8 * 128 * 128 * 128 * 2);
  _Float16* q1b   = (_Float16*)carve((size_t)NWIN * 8 * 64 * 4 * 2);
  _Float16* k1b   = (_Float16*)carve((size_t)NWIN * 8 * 64 * 8 * 2);
  _Float16* v1b   = (_Float16*)carve((size_t)NWIN * 8 * 64 * 16 * 2);
  _Float16* q2b   = (_Float16*)carve((size_t)NWIN * 8 * 64 * 4 * 2);
  _Float16* k2b   = (_Float16*)carve((size_t)NWIN * 8 * 64 * 8 * 2);
  _Float16* v2b   = (_Float16*)carve((size_t)NWIN * 8 * 64 * 16 * 2);

  // 1) weight prep
  prep_conv_w<<<(128 * 256 * 9 + 255) / 256, 256, 0, stream>>>(w_ic1, wt1, 128);
  prep_conv_w<<<(128 * 256 * 9 + 255) / 256, 256, 0, stream>>>(w_ic2, wt2, 128);
  prep_conv_w<<<(256 * 256 * 9 + 255) / 256, 256, 0, stream>>>(w_cat, wtc, 256);
  prep_qkv_w<<<(224 * 128 + 255) / 256, 256, 0, stream>>>(wq1, bq1, wk1, bk1, wv1,
                                                          bv1, wqkvA, bqkvA);
  prep_qkv_w<<<(224 * 128 + 255) / 256, 256, 0, stream>>>(wq2, bq2, wk2, bk2, wv2,
                                                          bv2, wqkvB, bqkvB);

  // 2) init 3x3 convs + BN + ReLU -> f16 activations
  conv3x3_kernel<true><<<dim3(BATCH * IMG, 1, 1), 256, 0, stream>>>(
      x1, x1, 256, wt1, g_ic1, b_ic1, x1f, 128);
  conv3x3_kernel<true><<<dim3(BATCH * IMG, 1, 1), 256, 0, stream>>>(
      x2, x2, 256, wt2, g_ic2, b_ic2, x2f, 128);

  // 3) fused q/k/v projections per window
  qkv_kernel<<<NWIN, 256, 0, stream>>>(x1f, wqkvA, bqkvA, q1b, k1b, v1b);
  qkv_kernel<<<NWIN, 256, 0, stream>>>(x2f, wqkvB, bqkvB, q2b, k2b, v2b);

  // 4) windowed cross attention -> x1_i / x2_i (f32, directly into d_out)
  attn_kernel<<<dim3(NWIN, NHEADS), 128, 0, stream>>>(q1b, k1b, v1b, q2b, k2b, v2b,
                                                      out1, out2);

  // 5) final 3x3 conv over concat(x1_i, x2_i) + BN + ReLU -> x_cat
  conv3x3_kernel<false><<<dim3(BATCH * IMG, 1, 2), 256, 0, stream>>>(
      out1, out2, 128, wtc, g_cat, b_cat, out0, 256);
}
